// WordGlobalAttention_63934883168399
// MI455X (gfx1250) — matrix operand order repaired and checked by reading, not compiled, not run on hardware
//
#include <hip/hip_runtime.h>
#include <math.h>

// Shapes from the reference
#define B_  32
#define N_  32      // S_NUM
#define S_  128     // S_LEN
#define D_  1024    // DIM
#define D2_ 2048

typedef __attribute__((ext_vector_type(16))) _Float16 v16h;
typedef __attribute__((ext_vector_type(8)))  float    v8f;

// ---------------------------------------------------------------------------
// Kernel 0: zero the per-batch context accumulator (d_ws is poisoned 0xAA)
// ---------------------------------------------------------------------------
__global__ void zero_c_kernel(float* __restrict__ c) {
    c[blockIdx.x * blockDim.x + threadIdx.x] = 0.0f;   // grid covers B_*D_ exactly
}

// ---------------------------------------------------------------------------
// Fragment packing helpers for V_WMMA_F32_16X16X32_F16 (wave32).
// A (16x32, MxK) 16-bit layout (ISA 7.12.2):
//   lane = h*16 + m ; VGPR r<4 : K = 8*h + 2*r + {0,1}; r>=4 : K = 16+8*h+2*(r-4)+{0,1}
// B (32x16, KxN): lane = kg*16 + n ; kernels pack A and B alike: element e holds K = kg*16 + e
// C/D (16x16 f32): VGPR r, lane l -> M = r + 8*(l>>4), N = l&15
// ---------------------------------------------------------------------------
__device__ __forceinline__ int a_frag_k(int e, int h) {
    const int r = e >> 1, p = e & 1;
    return (r < 4) ? (8 * h + 2 * r + p) : (16 + 8 * h + 2 * (r - 4) + p);
}

// ---------------------------------------------------------------------------
// Kernel 1: q = source @ W_in^T   [32x1024] = [32x1024] x [1024x1024]
// ---------------------------------------------------------------------------
__global__ __launch_bounds__(256) void qgemm_wmma(const float* __restrict__ src,
                                                  const float* __restrict__ Win,
                                                  float* __restrict__ q) {
    const int lane = threadIdx.x & 31;
    const int wave = threadIdx.x >> 5;
    const int tile = blockIdx.x * 8 + wave;       // 0..127
    const int row0 = (tile >> 6) * 16;            // M tile (batch rows)
    const int col0 = (tile & 63) * 16;            // N tile (q columns)
    const int m = lane & 15, h = lane >> 4;

    v8f acc = {};
    for (int k0 = 0; k0 < D_; k0 += 32) {
        v16h a, b;
#pragma unroll
        for (int e = 0; e < 16; ++e)
            a[e] = (_Float16)src[(row0 + m) * D_ + k0 + h * 16 + e];
#pragma unroll
        for (int e = 0; e < 16; ++e)              // B[k,n] = Win[col0+n, k0+k]
            b[e] = (_Float16)Win[(col0 + m) * D_ + k0 + h * 16 + e];
        acc = __builtin_amdgcn_wmma_f32_16x16x32_f16(false, a, false, b,
                                                     (short)0, acc, false, false);
    }
#pragma unroll
    for (int r = 0; r < 8; ++r)
        q[(row0 + r + 8 * h) * D_ + col0 + m] = acc[r];
}

// ---------------------------------------------------------------------------
// Async global -> LDS copy of one 512-float chunk (2 KB): 4 ops x 32 lanes x 16B.
// VDST VGPR holds the wave-relative LDS byte address (= low 32 bits of the
// generic pointer, per flat-address truncation rule). Tracked by ASYNCcnt.
// ---------------------------------------------------------------------------
__device__ __forceinline__ void async_chunk(const float* __restrict__ g,
                                            float* __restrict__ lbuf, int lane) {
#pragma unroll
    for (int j = 0; j < 4; ++j) {
        unsigned ldst = (unsigned)(uintptr_t)(lbuf + j * 128 + lane * 4);
        const float* gsrc = g + j * 128 + lane * 4;
        asm volatile("global_load_async_to_lds_b128 %0, %1, off"
                     :: "v"(ldst), "v"(gsrc)
                     : "memory");
    }
}

// ---------------------------------------------------------------------------
// Kernel 2: fused scores -> masked softmax -> align -> weighted sum.
// One workgroup (256 thr, 8 waves) per (b,n) sentence slab [128 x 1024] f32.
// Phase A streams the slab via double-buffered async-to-LDS (ASYNCcnt pipe);
// Phase C re-reads the slab (L2-resident) with direct float4 loads + atomics.
// ---------------------------------------------------------------------------
__global__ __launch_bounds__(256) void attn_fused(const float* __restrict__ q,
                                                  const float* __restrict__ mem,
                                                  const int*   __restrict__ mlens,
                                                  const float* __restrict__ salign,
                                                  const int*   __restrict__ snums,
                                                  float* __restrict__ align_out,
                                                  float* __restrict__ c_acc) {
    const int b = blockIdx.x >> 5;
    const int n = blockIdx.x & 31;
    const float* Mb = mem + ((size_t)(b * N_ + n)) * (size_t)(S_ * D_);

    __shared__ float s_q[D_];
    __shared__ float s_sc[S_];
    __shared__ float s_al[S_];
    __shared__ float s_stage[8 * 2 * 512];        // 8 waves x double-buffer x 2KB

    const int t = threadIdx.x, lane = t & 31, wave = t >> 5;

    // stage q[b,:] into LDS (256 threads x float4)
    ((float4*)s_q)[t] = ((const float4*)(q + (size_t)b * D_))[t];
    __syncthreads();

    // ---- Phase A: scores[s] = q . M[s,:]; wave handles rows s = wave + 8*i.
    // 32 half-row chunks per wave, double-buffered through wave-private LDS.
    {
        float* bufA = s_stage + wave * 1024;      // chunk buffer 0 (512 floats)
        float* bufB = bufA + 512;                 // chunk buffer 1
        const float4* Qr = (const float4*)s_q;
        float acc = 0.0f;
        async_chunk(Mb + (size_t)wave * D_, bufA, lane);      // prime chunk 0
        for (int kk = 0; kk < 32; ++kk) {
            float* cur = (kk & 1) ? bufB : bufA;
            if (kk < 31) {
                const int nk = kk + 1;
                const int ns = wave + 8 * (nk >> 1);
                async_chunk(Mb + (size_t)ns * D_ + (nk & 1) * 512,
                            (nk & 1) ? bufB : bufA, lane);
                asm volatile("s_wait_asynccnt 0x4" ::: "memory");
            } else {
                asm volatile("s_wait_asynccnt 0x0" ::: "memory");
            }
            const float4* Mr = (const float4*)cur;
            const int qb = (kk & 1) * 128;        // float4 offset of this half
#pragma unroll
            for (int c2 = 0; c2 < 4; ++c2) {
                float4 mv = Mr[c2 * 32 + lane];
                float4 qv = Qr[qb + c2 * 32 + lane];
                acc += mv.x * qv.x + mv.y * qv.y + mv.z * qv.z + mv.w * qv.w;
            }
            if (kk & 1) {                          // row finished -> reduce
                const int s = wave + 8 * (kk >> 1);
                float r = acc;
#pragma unroll
                for (int off = 16; off; off >>= 1) r += __shfl_xor(r, off, 32);
                if (lane == 0) s_sc[s] = r;
                acc = 0.0f;
            }
        }
    }
    __syncthreads();

    // ---- Phase B: masked softmax * sent_align (wave 0, 4 words per lane)
    if (wave == 0) {
        const int   len = mlens[b * N_ + n];
        const int   sv  = (n < snums[b]) ? 1 : 0;
        const float sa  = salign[b * N_ + n];
        float x[4], mk[4];
#pragma unroll
        for (int j = 0; j < 4; ++j) {
            const int s  = lane + 32 * j;
            const int ok = (s < len) && sv;
            mk[j] = ok ? 1.0f : 0.0f;
            x[j]  = ok ? s_sc[s] : -1e9f;
        }
        float mx = fmaxf(fmaxf(x[0], x[1]), fmaxf(x[2], x[3]));
#pragma unroll
        for (int off = 16; off; off >>= 1) mx = fmaxf(mx, __shfl_xor(mx, off, 32));
        float e[4], sum = 0.0f;
#pragma unroll
        for (int j = 0; j < 4; ++j) { e[j] = __expf(x[j] - mx); sum += e[j]; }
#pragma unroll
        for (int off = 16; off; off >>= 1) sum += __shfl_xor(sum, off, 32);
        const float scale = sa / sum;
#pragma unroll
        for (int j = 0; j < 4; ++j) {
            const int s = lane + 32 * j;
            const float al = e[j] * scale * mk[j];   // exact zeros where masked
            s_al[s] = al;
            align_out[(size_t)(b * N_ + n) * S_ + s] = al;
        }
    }
    __syncthreads();

    // ---- Phase C: c[b,d] += sum_s align[s] * M[s,d]  (slab hot in L2)
    float4 cacc = {0.f, 0.f, 0.f, 0.f};
    const int d = t * 4;
    for (int s = 0; s < S_; ++s) {
        const float a = s_al[s];
        const float4 mv = *(const float4*)(Mb + (size_t)s * D_ + d);
        cacc.x += a * mv.x; cacc.y += a * mv.y;
        cacc.z += a * mv.z; cacc.w += a * mv.w;
    }
    float* cp = c_acc + (size_t)b * D_ + d;
    atomicAdd(cp + 0, cacc.x); atomicAdd(cp + 1, cacc.y);
    atomicAdd(cp + 2, cacc.z); atomicAdd(cp + 3, cacc.w);
}

// ---------------------------------------------------------------------------
// Kernel 3: attn_h = tanh([c ; source] @ W_out^T)   [32x1024], K = 2048
// ---------------------------------------------------------------------------
__global__ __launch_bounds__(256) void outgemm_wmma(const float* __restrict__ c,
                                                    const float* __restrict__ src,
                                                    const float* __restrict__ Wout,
                                                    float* __restrict__ attn_h) {
    const int lane = threadIdx.x & 31;
    const int wave = threadIdx.x >> 5;
    const int tile = blockIdx.x * 8 + wave;       // 0..127
    const int row0 = (tile >> 6) * 16;
    const int col0 = (tile & 63) * 16;
    const int m = lane & 15, h = lane >> 4;

    v8f acc = {};
    for (int k0 = 0; k0 < D2_; k0 += 32) {
        const float* Abase = (k0 < D_) ? c : src;      // concat([c, source])
        const int kb = (k0 < D_) ? k0 : (k0 - D_);     // 32-window never straddles
        v16h a, b;
#pragma unroll
        for (int e = 0; e < 16; ++e)
            a[e] = (_Float16)Abase[(row0 + m) * D_ + kb + h * 16 + e];
#pragma unroll
        for (int e = 0; e < 16; ++e)
            b[e] = (_Float16)Wout[(col0 + m) * D2_ + k0 + h * 16 + e];
        acc = __builtin_amdgcn_wmma_f32_16x16x32_f16(false, a, false, b,
                                                     (short)0, acc, false, false);
    }
#pragma unroll
    for (int r = 0; r < 8; ++r)
        attn_h[(row0 + r + 8 * h) * D_ + col0 + m] = tanhf(acc[r]);
}

// ---------------------------------------------------------------------------
extern "C" void kernel_launch(void* const* d_in, const int* in_sizes, int n_in,
                              void* d_out, int out_size, void* d_ws, size_t ws_size,
                              hipStream_t stream) {
    const float* source         = (const float*)d_in[0];
    const float* memory_bank    = (const float*)d_in[1];
    const int*   memory_lengths = (const int*)  d_in[2];
    const float* sent_align     = (const float*)d_in[3];
    const int*   sent_nums      = (const int*)  d_in[4];
    const float* W_in           = (const float*)d_in[5];
    const float* W_out          = (const float*)d_in[6];

    float* attn_h    = (float*)d_out;              // [32,1024]
    float* align_out = (float*)d_out + B_ * D_;    // [32,32,128]

    float* q = (float*)d_ws;                       // [32,1024]
    float* c = q + B_ * D_;                        // [32,1024]

    zero_c_kernel<<<(B_ * D_) / 256, 256, 0, stream>>>(c);
    qgemm_wmma   <<<16, 256, 0, stream>>>(source, W_in, q);
    attn_fused   <<<B_ * N_, 256, 0, stream>>>(q, memory_bank, memory_lengths,
                                               sent_align, sent_nums, align_out, c);
    outgemm_wmma <<<16, 256, 0, stream>>>(c, source, W_out, attn_h);
}
